// JTMPN_16389595201591
// MI455X (gfx1250) — matrix operand order repaired and checked by reading, not compiled, not run on hardware
//
#include <hip/hip_runtime.h>
#include <hip/hip_bf16.h>

#define HIDDEN 256
#define ATOM_FDIM 35
#define BOND_FDIM 5
#define BOND_IN (ATOM_FDIM + BOND_FDIM)   // 40
#define OUT_IN  (ATOM_FDIM + HIDDEN)      // 291
#define MAX_NB 10
#define N_ATOMS 50000
#define N_BONDS 100000
#define N_MESS 10000
#define N_PAIRS 10000
#define N_MOLS 2000
#define ATOMS_PER_MOL 25

typedef __bf16 bf16;
typedef __attribute__((ext_vector_type(16))) __bf16 v16bf;
typedef __attribute__((ext_vector_type(8)))  __bf16 v8bf;
typedef __attribute__((ext_vector_type(8)))  float  v8f;
typedef __attribute__((ext_vector_type(4)))  float  v4f;

union Frag { v16bf v; v8bf h[2]; };

// ---------------------------------------------------------------------------
// K0: convert tree_message, W_h, W_o(hidden part) to bf16 tables in workspace
// ---------------------------------------------------------------------------
__global__ void k_convert(const float* __restrict__ tree,
                          const float* __restrict__ W_h,
                          const float* __restrict__ W_o,
                          bf16* __restrict__ tree_bf,
                          bf16* __restrict__ W_h_bf,
                          bf16* __restrict__ W_o_bf) {
  const int tid = blockIdx.x * blockDim.x + threadIdx.x;
  const int stride = gridDim.x * blockDim.x;
  const int TREE = N_MESS * HIDDEN;
  const int WH = HIDDEN * HIDDEN;
  for (int i = tid; i < TREE; i += stride) tree_bf[i] = (bf16)tree[i];
  for (int i = tid; i < WH; i += stride) {
    W_h_bf[i] = (bf16)W_h[i];
    int n = i / HIDDEN, k = i - n * HIDDEN;
    W_o_bf[i] = (bf16)W_o[(size_t)n * OUT_IN + ATOM_FDIM + k];
  }
}

// ---------------------------------------------------------------------------
// K1: binput = fbonds @ W_i^T  (K=40, memory bound) ; graph0 = relu -> bf16
// ---------------------------------------------------------------------------
__global__ void __launch_bounds__(HIDDEN)
k_binput(const float* __restrict__ fbonds, const float* __restrict__ W_i,
         float* __restrict__ binput, bf16* __restrict__ graph0) {
  __shared__ float fb[BOND_IN];
  const int b = blockIdx.x;
  const int n = threadIdx.x;
  if (n < BOND_IN) fb[n] = fbonds[(size_t)b * BOND_IN + n];
  __syncthreads();
  const float* w = &W_i[(size_t)n * BOND_IN];
  float acc = 0.f;
#pragma unroll
  for (int k = 0; k < BOND_IN; ++k) acc += fb[k] * w[k];
  const size_t o = (size_t)b * HIDDEN + n;
  binput[o] = acc;
  graph0[o] = (bf16)(acc > 0.f ? acc : 0.f);
}

// ---------------------------------------------------------------------------
// Shared device helper: gather-sum MAX_NB neighbor rows (bf16, f32 accum)
// into a 16x256 bf16 LDS tile. 128 threads: 16 rows x 8 col-chunks of 32.
// ---------------------------------------------------------------------------
__device__ __forceinline__
void gather_tile(const int* __restrict__ graph_row_base,  // &graph[(row0)*MAX_NB]
                 const bf16* __restrict__ tree_bf,
                 const bf16* __restrict__ msg_bf,
                 bf16 (*tileA)[HIDDEN + 8], int t) {
  const int rl = t >> 3;            // local row 0..15
  const int h0 = (t & 7) * 32;      // 32 hidden cols per thread
  float acc[32];
#pragma unroll
  for (int i = 0; i < 32; ++i) acc[i] = 0.f;
  const int* row = graph_row_base + (size_t)rl * MAX_NB;
#pragma unroll
  for (int j = 0; j < MAX_NB; ++j) {
    const int idx = row[j];
    const bf16* src = (idx < N_MESS)
                          ? (tree_bf + (size_t)idx * HIDDEN)
                          : (msg_bf + (size_t)(idx - N_MESS) * HIDDEN);
#pragma unroll
    for (int u = 0; u < 4; ++u) {
      v8bf v = *(const v8bf*)(src + h0 + u * 8);
#pragma unroll
      for (int e = 0; e < 8; ++e) acc[u * 8 + e] += (float)v[e];
    }
  }
#pragma unroll
  for (int i = 0; i < 32; ++i) tileA[rl][h0 + i] = (bf16)acc[i];
}

// ---------------------------------------------------------------------------
// K2: fused neighbor gather-sum + (nei @ W_h^T) via WMMA bf16 + binput + relu
//     block: 16 bonds x 256 outputs, 128 threads (4 waves x 4 n-tiles)
//     A fragments hoisted to registers; 8 B fragments preloaded per tile;
//     epilogue transposes C through per-wave LDS staging so binput loads and
//     bf16 stores are coalesced b128 accesses.
// ---------------------------------------------------------------------------
__global__ void __launch_bounds__(128)
k_mp(const int* __restrict__ bgraph,
     const bf16* __restrict__ tree_bf,
     const bf16* __restrict__ graph_src,
     const float* __restrict__ binput,
     const bf16* __restrict__ W_h_bf,
     bf16* __restrict__ graph_dst) {
  __shared__ bf16 tileA[16][HIDDEN + 8];      // padded to dodge bank conflicts
  __shared__ float stage[4][16][20];          // per-wave C transpose staging
  const int bond0 = blockIdx.x * 16;
  const int t = threadIdx.x;
  const int lane = t & 31;
  const int wid = t >> 5;

  __builtin_prefetch(&W_h_bf[(size_t)(wid * 64) * HIDDEN], 0, 1);

  gather_tile(&bgraph[(size_t)bond0 * MAX_NB], tree_bf, graph_src, tileA, t);
  __syncthreads();

  const int m = lane & 15;
  const int hi = lane >> 4;
  const int rr = lane >> 1;           // epilogue: row this lane stores
  const int cc = (lane & 1) * 8;      // epilogue: first of 8 contiguous cols

  // Hoist A fragments: same 16x256 tile feeds all 4 n-tiles of this wave.
  Frag a[HIDDEN / 32];
#pragma unroll
  for (int kk = 0; kk < HIDDEN / 32; ++kk) {
    const int ka = kk * 32 + hi * 16; // lanes0-15: K{0..15}; lanes16-31: K{16..31}
    a[kk].h[0] = *(const v8bf*)&tileA[m][ka];
    a[kk].h[1] = *(const v8bf*)&tileA[m][ka + 8];
  }

#pragma unroll
  for (int nt = 0; nt < 4; ++nt) {
    const int n0 = wid * 64 + nt * 16;
    const int n = n0 + m;               // this lane's B column / C column
    // Preload all 8 B fragments (one clause, one wait, then 8 chained WMMAs)
    Frag b[HIDDEN / 32];
#pragma unroll
    for (int kk = 0; kk < HIDDEN / 32; ++kk) {
      const int kb = kk * 32 + hi * 16; // per-lane contiguous 16-K run of row n
      b[kk].v = *(const v16bf*)&W_h_bf[(size_t)n * HIDDEN + kb];
    }
    v8f acc = {};
#pragma unroll
    for (int kk = 0; kk < HIDDEN / 32; ++kk)
      acc = __builtin_amdgcn_wmma_f32_16x16x32_bf16(false, a[kk].v, false, b[kk].v,
                                                    (short)0, acc, false, false);

    // --- epilogue: transpose C via per-wave LDS, then vectorized add+relu ---
#pragma unroll
    for (int r = 0; r < 8; ++r)         // C layout: M = r + 8*hi, N = n
      stage[wid][r + hi * 8][m] = acc[r];
    // same-wave DS ordering makes this safe without a barrier
    v4f s0 = *(const v4f*)&stage[wid][rr][cc];
    v4f s1 = *(const v4f*)&stage[wid][rr][cc + 4];
    const size_t o = (size_t)(bond0 + rr) * HIDDEN + n0 + cc;
    v4f b0 = *(const v4f*)&binput[o];
    v4f b1 = *(const v4f*)&binput[o + 4];
    v8bf outv;
#pragma unroll
    for (int e = 0; e < 4; ++e) {
      float x0 = s0[e] + b0[e]; x0 = x0 > 0.f ? x0 : 0.f;
      float x1 = s1[e] + b1[e]; x1 = x1 > 0.f ? x1 : 0.f;
      outv[e] = (bf16)x0;
      outv[e + 4] = (bf16)x1;
    }
    *(v8bf*)&graph_dst[o] = outv;       // one 16B store per lane, 512B per wave
  }
}

// ---------------------------------------------------------------------------
// K3: fused atom gather-sum + (concat(fatoms, nei) @ W_o^T + b_o) + relu
//     fatoms 35-dim part + bias added in C layout (W_o row reused 8x/lane),
//     then the same LDS transpose for coalesced f32 b128 stores.
// ---------------------------------------------------------------------------
__global__ void __launch_bounds__(128)
k_atoms(const int* __restrict__ agraph,
        const bf16* __restrict__ tree_bf,
        const bf16* __restrict__ graph_fin,
        const float* __restrict__ fatoms,
        const float* __restrict__ W_o,
        const float* __restrict__ b_o,
        const bf16* __restrict__ W_o_bf,
        float* __restrict__ atom_h) {
  __shared__ bf16 tileA[16][HIDDEN + 8];
  __shared__ float fat[16][ATOM_FDIM + 1];
  __shared__ float stage[4][16][20];
  const int atom0 = blockIdx.x * 16;
  const int t = threadIdx.x;
  const int lane = t & 31;
  const int wid = t >> 5;

  for (int i = t; i < 16 * ATOM_FDIM; i += 128) {
    const int a = i / ATOM_FDIM, k = i - a * ATOM_FDIM;
    fat[a][k] = fatoms[(size_t)(atom0 + a) * ATOM_FDIM + k];
  }
  gather_tile(&agraph[(size_t)atom0 * MAX_NB], tree_bf, graph_fin, tileA, t);
  __syncthreads();

  const int m = lane & 15;
  const int hi = lane >> 4;
  const int rr = lane >> 1;
  const int cc = (lane & 1) * 8;

  Frag a[HIDDEN / 32];
#pragma unroll
  for (int kk = 0; kk < HIDDEN / 32; ++kk) {
    const int ka = kk * 32 + hi * 16;
    a[kk].h[0] = *(const v8bf*)&tileA[m][ka];
    a[kk].h[1] = *(const v8bf*)&tileA[m][ka + 8];
  }

#pragma unroll
  for (int nt = 0; nt < 4; ++nt) {
    const int n0 = wid * 64 + nt * 16;
    const int n = n0 + m;
    Frag b[HIDDEN / 32];
#pragma unroll
    for (int kk = 0; kk < HIDDEN / 32; ++kk) {
      const int kb = kk * 32 + hi * 16;
      b[kk].v = *(const v16bf*)&W_o_bf[(size_t)n * HIDDEN + kb];
    }
    v8f acc = {};
#pragma unroll
    for (int kk = 0; kk < HIDDEN / 32; ++kk)
      acc = __builtin_amdgcn_wmma_f32_16x16x32_bf16(false, a[kk].v, false, b[kk].v,
                                                    (short)0, acc, false, false);

    const float* wrow = &W_o[(size_t)n * OUT_IN];   // fatoms part (K = 0..34), f32
    const float bias = b_o[n];
#pragma unroll
    for (int r = 0; r < 8; ++r) {
      const int mm = r + hi * 8;
      float v = acc[r] + bias;
#pragma unroll
      for (int k = 0; k < ATOM_FDIM; ++k) v += fat[mm][k] * wrow[k];
      stage[wid][mm][m] = v;
    }
    v4f s0 = *(const v4f*)&stage[wid][rr][cc];
    v4f s1 = *(const v4f*)&stage[wid][rr][cc + 4];
#pragma unroll
    for (int e = 0; e < 4; ++e) {
      s0[e] = s0[e] > 0.f ? s0[e] : 0.f;
      s1[e] = s1[e] > 0.f ? s1[e] : 0.f;
    }
    const size_t o = (size_t)(atom0 + rr) * HIDDEN + n0 + cc;
    *(v4f*)&atom_h[o] = s0;
    *(v4f*)&atom_h[o + 4] = s1;
  }
}

// ---------------------------------------------------------------------------
// K4: mol mean pooling over 25 atoms -> d_out[0 .. N_MOLS*HIDDEN)
// ---------------------------------------------------------------------------
__global__ void __launch_bounds__(HIDDEN)
k_pool(const float* __restrict__ atom_h, float* __restrict__ out) {
  const int mol = blockIdx.x;
  const int h = threadIdx.x;
  float s = 0.f;
  const float* base = &atom_h[(size_t)mol * ATOMS_PER_MOL * HIDDEN + h];
#pragma unroll
  for (int a = 0; a < ATOMS_PER_MOL; ++a) s += base[(size_t)a * HIDDEN];
  out[(size_t)mol * HIDDEN + h] = s * (1.0f / ATOMS_PER_MOL);
}

// ---------------------------------------------------------------------------
// K5: collated = graph_message[pair_idx].mean(axis=1)
// ---------------------------------------------------------------------------
__global__ void __launch_bounds__(HIDDEN)
k_collated(const int* __restrict__ pair_idx, const bf16* __restrict__ graph_fin,
           float* __restrict__ out) {
  const int p = blockIdx.x;
  const int h = threadIdx.x;
  const int i0 = pair_idx[p * 2 + 0];
  const int i1 = pair_idx[p * 2 + 1];
  const float v = 0.5f * ((float)graph_fin[(size_t)i0 * HIDDEN + h] +
                          (float)graph_fin[(size_t)i1 * HIDDEN + h]);
  out[(size_t)p * HIDDEN + h] = v;
}

// ---------------------------------------------------------------------------
extern "C" void kernel_launch(void* const* d_in, const int* in_sizes, int n_in,
                              void* d_out, int out_size, void* d_ws, size_t ws_size,
                              hipStream_t stream) {
  (void)in_sizes; (void)n_in; (void)out_size; (void)ws_size;
  const float* fatoms   = (const float*)d_in[0];
  const float* fbonds   = (const float*)d_in[1];
  const float* tree     = (const float*)d_in[2];
  const int*   agraph   = (const int*)d_in[3];
  const int*   bgraph   = (const int*)d_in[4];
  const int*   pair_idx = (const int*)d_in[5];
  const float* W_i      = (const float*)d_in[6];
  const float* W_h      = (const float*)d_in[7];
  const float* W_o      = (const float*)d_in[8];
  const float* b_o      = (const float*)d_in[9];
  float* out = (float*)d_out;

  char* ws = (char*)d_ws;
  size_t off = 0;
  auto take = [&](size_t bytes) -> void* {
    void* p = ws + off;
    off += (bytes + 255) & ~(size_t)255;
    return p;
  };
  bf16*  tree_bf = (bf16*)take((size_t)N_MESS  * HIDDEN * sizeof(bf16));
  bf16*  graphA  = (bf16*)take((size_t)N_BONDS * HIDDEN * sizeof(bf16));
  bf16*  graphB  = (bf16*)take((size_t)N_BONDS * HIDDEN * sizeof(bf16));
  float* binput  = (float*)take((size_t)N_BONDS * HIDDEN * sizeof(float));
  bf16*  W_h_bf  = (bf16*)take((size_t)HIDDEN * HIDDEN * sizeof(bf16));
  bf16*  W_o_bf  = (bf16*)take((size_t)HIDDEN * HIDDEN * sizeof(bf16));
  float* atom_h  = (float*)take((size_t)N_ATOMS * HIDDEN * sizeof(float));

  k_convert<<<2048, 256, 0, stream>>>(tree, W_h, W_o, tree_bf, W_h_bf, W_o_bf);
  k_binput<<<N_BONDS, HIDDEN, 0, stream>>>(fbonds, W_i, binput, graphA);

  // DEPTH-1 = 3 message-passing iterations, ping-pong A -> B -> A -> B
  k_mp<<<N_BONDS / 16, 128, 0, stream>>>(bgraph, tree_bf, graphA, binput, W_h_bf, graphB);
  k_mp<<<N_BONDS / 16, 128, 0, stream>>>(bgraph, tree_bf, graphB, binput, W_h_bf, graphA);
  k_mp<<<N_BONDS / 16, 128, 0, stream>>>(bgraph, tree_bf, graphA, binput, W_h_bf, graphB);
  const bf16* graph_fin = graphB;

  k_atoms<<<N_ATOMS / 16, 128, 0, stream>>>(agraph, tree_bf, graph_fin, fatoms,
                                            W_o, b_o, W_o_bf, atom_h);
  k_pool<<<N_MOLS, HIDDEN, 0, stream>>>(atom_h, out);
  k_collated<<<N_PAIRS, HIDDEN, 0, stream>>>(pair_idx, graph_fin,
                                             out + (size_t)N_MOLS * HIDDEN);
}
